// MultiHeadAttn_31404800868449
// MI455X (gfx1250) — matrix operand order repaired and checked by reading, not compiled, not run on hardware
//
#include <hip/hip_runtime.h>

// ---- problem constants (match reference) ----
#define NH   16
#define SQ   2048
#define DKv  64
#define WIN  256
#define QT   16     // query tile (per wave)
#define KT   32     // key tile per iteration
#define WPB  4      // waves per block (4 consecutive q-tiles, same head, shared K/V tiles)

// scale = 1/sqrt(64) folded with log2(e): softmax computed in base-2 domain
#define SCL2 0.18033688011112042f   // 0.125 * 1.4426950408889634

#if __has_builtin(__builtin_amdgcn_global_load_async_to_lds_b128) && \
    __has_builtin(__builtin_amdgcn_s_wait_asynccnt)
#define USE_ASYNC 1
#else
#define USE_ASYNC 0
#endif

typedef __attribute__((ext_vector_type(16))) _Float16 v16h;
typedef __attribute__((ext_vector_type(8)))  float    v8f;
typedef int v4i_vec __attribute__((vector_size(4 * sizeof(int))));   // matches builtin param type

union H16 {
    v16h     v;
    uint4    u4[2];
    unsigned u32[8];
    _Float16 h[16];
};

// two runs of 8 contiguous f16 from LDS (16B-aligned by construction) -> A-operand v16h
__device__ __forceinline__ v16h lds_ld8x2(const _Float16* p0, const _Float16* p1) {
    H16 r;
    r.u4[0] = *(const uint4*)(p0);
    r.u4[1] = *(const uint4*)(p1);
    return r.v;
}
// pack two f32 -> one dword of two f16 (v_cvt_pk_f16_f32)
__device__ __forceinline__ unsigned pk2h(float a, float b) {
    union { _Float16 h[2]; unsigned u; } t;
    t.h[0] = (_Float16)a;
    t.h[1] = (_Float16)b;
    return t.u;
}

#if USE_ASYNC
__device__ __forceinline__ void async_cp_b128(const float* g, float* l) {
    __builtin_amdgcn_global_load_async_to_lds_b128((v4i_vec*)g, (v4i_vec*)l, 0, 0);
}
#endif

__global__ __launch_bounds__(WPB * 32)
void attn_fa_kernel(const float* __restrict__ Q, const float* __restrict__ K,
                    const float* __restrict__ V, const int* __restrict__ layer_idx,
                    float* __restrict__ Out)
{
    // block-shared f16 tiles (all 4 waves read the same key block)
    __shared__ __align__(16) _Float16 kh[KT * DKv];   // [j][d], row stride 64 halves (128B)
    __shared__ __align__(16) _Float16 vt[DKv * KT];   // [d][j], row stride 32 halves (64B)
#if USE_ASYNC
    // double-buffered raw f32 staging: [buf][ K(2048 f32) | V(2048 f32) ]
    __shared__ __align__(16) float sStage[2][2 * KT * DKv];
#endif

    const int t       = threadIdx.x;        // 0..127
    const int lane    = t & 31;
    const int wave    = t >> 5;
    const int lane_lo = lane & 15;
    const int lane_hi = lane >> 4;          // 0 or 1
    const bool hi_half = (lane_hi != 0);

    const int tile = blockIdx.x * WPB + wave;   // 0..2047 (4 consecutive tiles, same head)
    const int h    = tile >> 7;                 // 128 q-tiles per head
    const int qb   = (tile & 127) << 4;         // this wave's query tile base
    const int qb0  = ((blockIdx.x * WPB) & 127) << 4;   // block's first q-tile base

    const bool local = (layer_idx[0] & 1) != 0;

    const float* kbase = K + (size_t)h * SQ * DKv;
    const float* vbase = V + (size_t)h * SQ * DKv;

    // ---- preload Q tile as B-operand (lane col = query m = lane_lo; elem e -> d = dc*32 + lane_hi*16 + e)
    v16h qop[2];
    {
        const float* qrow = Q + (size_t)(h * SQ + qb + lane_lo) * DKv;
        #pragma unroll
        for (int dc = 0; dc < 2; ++dc) {
            const int d0 = dc * 32 + lane_hi * 16;
            float4 f0 = *(const float4*)(qrow + d0);
            float4 f1 = *(const float4*)(qrow + d0 + 4);
            float4 f2 = *(const float4*)(qrow + d0 + 8);
            float4 f3 = *(const float4*)(qrow + d0 + 12);
            H16 a;
            a.u32[0] = pk2h(f0.x, f0.y); a.u32[1] = pk2h(f0.z, f0.w);
            a.u32[2] = pk2h(f1.x, f1.y); a.u32[3] = pk2h(f1.z, f1.w);
            a.u32[4] = pk2h(f2.x, f2.y); a.u32[5] = pk2h(f2.z, f2.w);
            a.u32[6] = pk2h(f3.x, f3.y); a.u32[7] = pk2h(f3.z, f3.w);
            qop[dc] = a.v;
        }
    }

    // ones A-operand for row-sum WMMA (l = 1^T * P^T, replicated into every row/lane)
    v16h onesv;
    {
        H16 o;
        #pragma unroll
        for (int i = 0; i < 16; ++i) o.h[i] = (_Float16)1.0f;
        onesv = o.v;
    }

    // ---- accumulators: O^T D-layout (vgpr r = d row within 16-chunk, lane col = query m)
    v8f acc[4], accl;
    #pragma unroll
    for (int dc = 0; dc < 4; ++dc)
        #pragma unroll
        for (int r = 0; r < 8; ++r) acc[dc][r] = 0.0f;
    #pragma unroll
    for (int r = 0; r < 8; ++r) accl[r] = 0.0f;

    float mrun = -1e30f;                    // per-lane running max (log2 domain)
    const int qi = qb + lane_lo;            // this lane's query index

    // ---- key-block ranges: union (block-uniform, drives fill+barriers) / own (wave-uniform compute)
    const int jeU = (qb0 + (WPB * QT - 1)) & ~(KT - 1);
    int jsU = 0;
    if (local) { int lo = qb0 - (WIN - 1); if (lo < 0) lo = 0; jsU = lo & ~(KT - 1); }
    const int jeO = (qb + QT - 1) & ~(KT - 1);
    int jsO = 0;
    if (local) { int lo = qb - (WIN - 1); if (lo < 0) lo = 0; jsO = lo & ~(KT - 1); }

#if USE_ASYNC
    // prologue: stage first block asynchronously (8 x b128 per thread, ASYNCcnt-tracked)
    {
        float* stg = sStage[0];
        #pragma unroll
        for (int i = 0; i < 4; ++i) {
            const int f4 = (i << 7) + t;    // float4 index 0..511
            async_cp_b128(kbase + (size_t)jsU * DKv + f4 * 4, stg + f4 * 4);
            async_cp_b128(vbase + (size_t)jsU * DKv + f4 * 4, stg + KT * DKv + f4 * 4);
        }
    }
    int cb = 0;
#endif

    for (int jb = jsU; jb <= jeU; jb += KT) {
        const bool have_next = (jb + KT <= jeU);

#if USE_ASYNC
        // issue next block into the other staging buffer, then wait for current (in-order retire)
        if (have_next) {
            float* stg = sStage[cb ^ 1];
            #pragma unroll
            for (int i = 0; i < 4; ++i) {
                const int f4 = (i << 7) + t;
                async_cp_b128(kbase + (size_t)(jb + KT) * DKv + f4 * 4, stg + f4 * 4);
                async_cp_b128(vbase + (size_t)(jb + KT) * DKv + f4 * 4, stg + KT * DKv + f4 * 4);
            }
            __builtin_amdgcn_s_wait_asynccnt(8);
        } else {
            __builtin_amdgcn_s_wait_asynccnt(0);
        }
#endif
        // all copies visible + previous compute finished reading the f16 tiles
        __syncthreads();

        // ---- convert current block -> shared f16 tiles (work split across all 128 threads) ----
        {
#if USE_ASYNC
            const float* ksrc = sStage[cb];
            const float* vsrc = sStage[cb] + KT * DKv;
#else
            const float* ksrc = kbase + (size_t)jb * DKv;
            const float* vsrc = vbase + (size_t)jb * DKv;
            if (have_next) {
                __builtin_prefetch(kbase + (size_t)(jb + KT) * DKv + t * 16, 0, 1);
                __builtin_prefetch(vbase + (size_t)(jb + KT) * DKv + t * 16, 0, 1);
            }
#endif
            #pragma unroll
            for (int i = 0; i < 4; ++i) {
                const int f4 = (i << 7) + t;        // float4 index 0..511
                const int j  = f4 >> 4;             // key row 0..31
                const int dp = (f4 & 15) << 2;      // d position 0,4,..,60
                float4 kf = *(const float4*)(ksrc + f4 * 4);
                float4 vf = *(const float4*)(vsrc + f4 * 4);
                _Float16* kd = kh + j * DKv + dp;
                kd[0] = (_Float16)kf.x; kd[1] = (_Float16)kf.y;
                kd[2] = (_Float16)kf.z; kd[3] = (_Float16)kf.w;
                vt[(dp + 0) * KT + j] = (_Float16)vf.x;
                vt[(dp + 1) * KT + j] = (_Float16)vf.y;
                vt[(dp + 2) * KT + j] = (_Float16)vf.z;
                vt[(dp + 3) * KT + j] = (_Float16)vf.w;
            }
        }
        __syncthreads();
#if USE_ASYNC
        cb ^= 1;
#endif

        // ---- wave-uniform guard: this block visible to this wave's queries? (EXEC stays full)
        if (jb < jsO || jb > jeO) continue;

        // ---- scores transposed: S^T = K * Q^T  (D rows = keys, cols = queries) ----
        v8f s0, s1;
        #pragma unroll
        for (int r = 0; r < 8; ++r) { s0[r] = 0.0f; s1[r] = 0.0f; }
        #pragma unroll
        for (int dc = 0; dc < 2; ++dc) {
            const int doff = dc * 32 + lane_hi * 16;
            v16h a0 = lds_ld8x2(kh + (lane_lo)      * DKv + doff,
                                kh + (lane_lo)      * DKv + doff + 8);
            v16h a1 = lds_ld8x2(kh + (16 + lane_lo) * DKv + doff,
                                kh + (16 + lane_lo) * DKv + doff + 8);
            s0 = __builtin_amdgcn_wmma_f32_16x16x32_f16(false, a0, false, qop[dc],
                                                        (short)0, s0, false, false);
            s1 = __builtin_amdgcn_wmma_f32_16x16x32_f16(false, a1, false, qop[dc],
                                                        (short)0, s1, false, false);
        }

        // ---- per-lane online softmax (query m = lane_lo; 16 key values in registers) ----
        float x0[8], x1[8];
        bool  g0[8], g1[8];
        #pragma unroll
        for (int r = 0; r < 8; ++r) {
            const int kj0 = jb + lane_hi * 8 + r;
            const int kj1 = kj0 + 16;
            g0[r] = (kj0 <= qi) && (!local || (qi - kj0) < WIN);
            g1[r] = (kj1 <= qi) && (!local || (qi - kj1) < WIN);
            x0[r] = g0[r] ? s0[r] * SCL2 : -1e30f;
            x1[r] = g1[r] ? s1[r] * SCL2 : -1e30f;
        }
        float rm = fmaxf(x0[0], x1[0]);
        #pragma unroll
        for (int r = 1; r < 8; ++r) rm = fmaxf(rm, fmaxf(x0[r], x1[r]));
        rm = fmaxf(rm, __shfl_xor(rm, 16, 32));       // combine the two lane halves of this query

        const float mnew  = fmaxf(mrun, rm);
        const float alpha = exp2f(mrun - mnew);
        mrun = mnew;

        float p0[8], p1[8];
        #pragma unroll
        for (int r = 0; r < 8; ++r) {
            p0[r] = g0[r] ? exp2f(x0[r] - mnew) : 0.0f;   // explicit zero: fully-masked rows stay clean
            p1[r] = g1[r] ? exp2f(x1[r] - mnew) : 0.0f;
        }

        // ---- build P^T B-operand in registers (lane col = query; elem e -> j = jb + lane_hi*16 + e)
        unsigned pk0[4], pk1[4], rcv[4];
        #pragma unroll
        for (int w = 0; w < 4; ++w) {
            pk0[w] = pk2h(p0[2 * w], p0[2 * w + 1]);
            pk1[w] = pk2h(p1[2 * w], p1[2 * w + 1]);
            unsigned snd = hi_half ? pk0[w] : pk1[w];     // send what the partner half needs
            rcv[w] = (unsigned)__shfl_xor((int)snd, 16, 32);
        }
        H16 pb;
        #pragma unroll
        for (int w = 0; w < 4; ++w) {
            pb.u32[w]     = hi_half ? rcv[w] : pk0[w];
            pb.u32[4 + w] = hi_half ? pk1[w] : rcv[w];
        }
        const v16h pbv = pb.v;

        // ---- rescale accumulators only when some lane's max actually moved ----
        if (__any(alpha < 1.0f)) {
            #pragma unroll
            for (int dc = 0; dc < 4; ++dc)
                #pragma unroll
                for (int r = 0; r < 8; ++r) acc[dc][r] *= alpha;
            #pragma unroll
            for (int r = 0; r < 8; ++r) accl[r] *= alpha;
        }

        // ---- O^T += V^T * P^T  (4 d-chunks) + row-sum via ones-WMMA ----
        #pragma unroll
        for (int dc = 0; dc < 4; ++dc) {
            const int joff = lane_hi * 16;
            v16h va = lds_ld8x2(vt + (dc * 16 + lane_lo) * KT + joff,
                                vt + (dc * 16 + lane_lo) * KT + joff + 8);
            acc[dc] = __builtin_amdgcn_wmma_f32_16x16x32_f16(false, va, false, pbv,
                                                             (short)0, acc[dc], false, false);
        }
        accl = __builtin_amdgcn_wmma_f32_16x16x32_f16(false, onesv, false, pbv,
                                                      (short)0, accl, false, false);
    }

    // ---- epilogue: normalize, store context (O^T: vgpr r -> d = dc*16 + lane_hi*8 + r; lane col = query)
    const float inv = 1.0f / accl[0];       // all rows of accl hold the same row-sum
    float* orow = Out + (size_t)(h * SQ + qb + lane_lo) * DKv + lane_hi * 8;
    #pragma unroll
    for (int dc = 0; dc < 4; ++dc) {
        float4 loq, hiq;
        loq.x = acc[dc][0] * inv; loq.y = acc[dc][1] * inv;
        loq.z = acc[dc][2] * inv; loq.w = acc[dc][3] * inv;
        hiq.x = acc[dc][4] * inv; hiq.y = acc[dc][5] * inv;
        hiq.z = acc[dc][6] * inv; hiq.w = acc[dc][7] * inv;
        *(float4*)(orow + dc * 16)     = loq;
        *(float4*)(orow + dc * 16 + 4) = hiq;
    }
}

// ---- passthrough: present = (k, v) ----
__global__ void kv_copy_kernel(const float4* __restrict__ k, const float4* __restrict__ v,
                               float4* __restrict__ outk, float4* __restrict__ outv, int n4)
{
    int i = blockIdx.x * blockDim.x + threadIdx.x;
    if (i < n4) {
        outk[i] = k[i];
        outv[i] = v[i];
    }
}

extern "C" void kernel_launch(void* const* d_in, const int* in_sizes, int n_in,
                              void* d_out, int out_size, void* d_ws, size_t ws_size,
                              hipStream_t stream)
{
    const float* q = (const float*)d_in[0];
    const float* k = (const float*)d_in[1];
    const float* v = (const float*)d_in[2];
    const int* layer_idx = (const int*)d_in[3];
    // d_in[4] = training (0) -> no dropout

    float* out  = (float*)d_out;
    const size_t per = (size_t)NH * SQ * DKv;   // 2,097,152 elements
    float* outk = out + per;
    float* outv = outk + per;

    const int tiles  = NH * (SQ / QT);          // 2048 wave-tiles
    const int blocks = tiles / WPB;             // 512
    attn_fa_kernel<<<blocks, WPB * 32, 0, stream>>>(q, k, v, layer_idx, out);

    const int n4 = (int)(per / 4);
    kv_copy_kernel<<<(n4 + 255) / 256, 256, 0, stream>>>(
        (const float4*)k, (const float4*)v, (float4*)outk, (float4*)outv, n4);
}
